// DynamicWeightProjection_37606733643840
// MI455X (gfx1250) — matrix operand order repaired and checked by reading, not compiled, not run on hardware
//
#include <hip/hip_runtime.h>
#include <hip/hip_bf16.h>
#include <math.h>

typedef __attribute__((ext_vector_type(16))) __bf16 v16bf;
typedef __attribute__((ext_vector_type(8)))  float  v8f;

union Frag {
  uint4 q[2];
  v16bf v;
};

__device__ __forceinline__ unsigned pack2_bf16(float a, float b) {
  __bf16 x = (__bf16)a;
  __bf16 y = (__bf16)b;
  unsigned short ux = __builtin_bit_cast(unsigned short, x);
  unsigned short uy = __builtin_bit_cast(unsigned short, y);
  return (unsigned)ux | ((unsigned)uy << 16);
}

// Problem geometry
#define ROWS   16384            // B*T
#define DK     4096             // D
#define NKT    128              // DK / 32
#define NT1    20               // 320 packed columns / 16
#define BTILE_DW (NT1 * 32 * 8)          // 5120 dwords = 20KB per K-step
#define BPACK_DW (NKT * BTILE_DW)        // 655360 dwords
#define QPACK_DW (2 * 4 * 8 * 32 * 8)    // 16384 dwords

// ---------------------------------------------------------------------------
// Pack dw1 (c in {0,2}) and dd (m in 0..31, 64..95) into WMMA bf16 B-fragment
// layout: dword index = ((kt*NT1 + nt)*32 + lane)*8 + j
//   col n = nt*16 + (lane&15);  k0 = kt*32 + (lane>>4)*16 + 2*j; pair (k0,k0+1)
// ---------------------------------------------------------------------------
__global__ void pack_b_kernel(const float* __restrict__ dw1,
                              const float* __restrict__ dd,
                              unsigned* __restrict__ bpack) {
  int idx = blockIdx.x * 256 + threadIdx.x;
  if (idx >= BPACK_DW) return;
  int j    = idx & 7;
  int t    = idx >> 3;
  int lane = t & 31;
  int t2   = t >> 5;
  int nt   = t2 % NT1;
  int kt   = t2 / NT1;
  int n  = nt * 16 + (lane & 15);
  int k0 = kt * 32 + (lane >> 4) * 16 + 2 * j;
  float v0, v1;
  if (n < 256) {
    int c  = (n >> 7) * 2;          // c = 0 or 2
    int kk = n & 127;
    v0 = dw1[(size_t)k0 * 512 + c * 128 + kk];
    v1 = dw1[(size_t)(k0 + 1) * 512 + c * 128 + kk];
  } else {
    int jj = n - 256;
    int m  = (jj < 32) ? jj : (jj + 32);   // m in 0..31 or 64..95
    v0 = dd[(size_t)k0 * 128 + m];
    v1 = dd[(size_t)(k0 + 1) * 128 + m];
  }
  bpack[idx] = pack2_bf16(v0, v1);
}

// ---------------------------------------------------------------------------
// Pack qkw (c in {0,2}) into stage-2 WMMA bf16 B-fragment layout:
// dword index = (((cp*4 + kt2)*8 + nt)*32 + lane)*8 + j
// ---------------------------------------------------------------------------
__global__ void pack_q_kernel(const float* __restrict__ qkw,
                              unsigned* __restrict__ qpack) {
  int idx = blockIdx.x * 256 + threadIdx.x;   // < QPACK_DW
  int j    = idx & 7;
  int lane = (idx >> 3) & 31;
  int nt   = (idx >> 8) & 7;
  int kt2  = (idx >> 11) & 3;
  int cp   = idx >> 13;
  int im = nt * 16 + (lane & 15);
  int k0 = kt2 * 32 + (lane >> 4) * 16 + 2 * j;
  int c  = cp * 2;
  float v0 = qkw[(size_t)c * 16384 + (size_t)k0 * 128 + im];
  float v1 = qkw[(size_t)c * 16384 + (size_t)(k0 + 1) * 128 + im];
  qpack[idx] = pack2_bf16(v0, v1);
}

// ---------------------------------------------------------------------------
// Main fused kernel: 256 blocks x 128 threads (4 waves). 64 rows per block,
// each wave owns 16 rows x 320 cols (20 WMMA accumulator tiles).
// Double-buffered LDS; B tiles staged with GLOBAL_LOAD_ASYNC_TO_LDS_B128.
// ---------------------------------------------------------------------------
struct S1 {
  unsigned a[2][64 * 20];      // A tiles bf16: 64 rows, pitch 20 dw (80B)
  unsigned b[2][BTILE_DW];     // B tiles: 20 fragment tiles each (20KB)
};
union LdsU {
  S1 s1;                       // 51200 B
  unsigned h[4][16 * 68];      // stage2: per-wave 16 rows x pitch 68 dw
};

__global__ __launch_bounds__(128, 1)
void dwp_main_kernel(const float* __restrict__ A,
                     const unsigned* __restrict__ bpack,
                     const unsigned* __restrict__ qpack,
                     const float* __restrict__ nscale,
                     float* __restrict__ out) {
  __shared__ LdsU lds;
  const int tid  = threadIdx.x;
  const int wave = tid >> 5;
  const int lane = tid & 31;
  const int rowBase = blockIdx.x * 64;

  const v8f vzero = {0.f, 0.f, 0.f, 0.f, 0.f, 0.f, 0.f, 0.f};
  v8f acc[NT1];
#pragma unroll
  for (int i = 0; i < NT1; ++i) acc[i] = vzero;

  // A-staging assignment: thread t handles row t>>1, 16-float half (t&1)
  const int arow  = tid >> 1;
  const int ahalf = tid & 1;
  const float4* aglob = (const float4*)(A + (size_t)(rowBase + arow) * DK + ahalf * 16);
  const int astOff = arow * 20 + ahalf * 8;

  // per-thread LDS byte addresses (low 32 bits of flat LDS address = LDS offset)
  const unsigned ldsB0 = (unsigned)(uintptr_t)&lds.s1.b[0][0] + (unsigned)tid * 16u;
  const unsigned ldsB1 = (unsigned)(uintptr_t)&lds.s1.b[1][0] + (unsigned)tid * 16u;
  const unsigned gOff  = (unsigned)tid * 16u;   // per-lane global byte offset

  // ---- prologue: stage buffers 0 ----
  {
    const float4* ap = aglob;                  // kt = 0
    float4 f0 = ap[0], f1 = ap[1], f2 = ap[2], f3 = ap[3];
    unsigned* as = &lds.s1.a[0][astOff];
    as[0] = pack2_bf16(f0.x, f0.y);  as[1] = pack2_bf16(f0.z, f0.w);
    as[2] = pack2_bf16(f1.x, f1.y);  as[3] = pack2_bf16(f1.z, f1.w);
    as[4] = pack2_bf16(f2.x, f2.y);  as[5] = pack2_bf16(f2.z, f2.w);
    as[6] = pack2_bf16(f3.x, f3.y);  as[7] = pack2_bf16(f3.z, f3.w);
    const unsigned* bbase = bpack;             // kt = 0 tile
#pragma unroll
    for (int i = 0; i < 10; ++i) {
      asm volatile("global_load_async_to_lds_b128 %0, %1, %2 offset:%3"
                   :: "v"(ldsB0), "v"(gOff), "s"(bbase), "i"(i * 2048)
                   : "memory");
    }
    asm volatile("s_wait_asynccnt 0x0" ::: "memory");
  }
  __syncthreads();

  for (int kt = 0; kt < NKT; ++kt) {
    const int cur = kt & 1;
    const int nxt = cur ^ 1;
    const bool pf = (kt + 1 < NKT);

    // ---- prefetch next iteration: async B -> LDS, A -> regs ----
    float4 f0, f1, f2, f3;
    if (pf) {
      const unsigned* bbase = bpack + (size_t)(kt + 1) * BTILE_DW;
      const unsigned ldsBn = nxt ? ldsB1 : ldsB0;
#pragma unroll
      for (int i = 0; i < 10; ++i) {
        asm volatile("global_load_async_to_lds_b128 %0, %1, %2 offset:%3"
                     :: "v"(ldsBn), "v"(gOff), "s"(bbase), "i"(i * 2048)
                     : "memory");
      }
      const float4* ap = aglob + (kt + 1) * 8;
      f0 = ap[0]; f1 = ap[1]; f2 = ap[2]; f3 = ap[3];
    }

    // ---- compute on current buffers ----
    Frag af;
    {
      const unsigned* ar = &lds.s1.a[cur][(wave * 16 + (lane & 15)) * 20 + ((lane >> 4) << 3)];
      af.q[0] = *(const uint4*)(ar);      // K 0..7  (or 16..23 for hi lanes)
      af.q[1] = *(const uint4*)(ar + 4);  // K 8..15 (or 24..31)
    }
#pragma unroll
    for (int nt = 0; nt < NT1; ++nt) {
      Frag bf;
      const unsigned* bb = &lds.s1.b[cur][(nt * 32 + lane) * 8];
      bf.q[0] = *(const uint4*)(bb);
      bf.q[1] = *(const uint4*)(bb + 4);
      acc[nt] = __builtin_amdgcn_wmma_f32_16x16x32_bf16(
          false, af.v, false, bf.v, (short)0, acc[nt], false, false);
    }

    // ---- convert+store next A tile, then close the pipeline stage ----
    if (pf) {
      unsigned* as = &lds.s1.a[nxt][astOff];
      as[0] = pack2_bf16(f0.x, f0.y);  as[1] = pack2_bf16(f0.z, f0.w);
      as[2] = pack2_bf16(f1.x, f1.y);  as[3] = pack2_bf16(f1.z, f1.w);
      as[4] = pack2_bf16(f2.x, f2.y);  as[5] = pack2_bf16(f2.z, f2.w);
      as[6] = pack2_bf16(f3.x, f3.y);  as[7] = pack2_bf16(f3.z, f3.w);
      asm volatile("s_wait_asynccnt 0x0" ::: "memory");
    }
    __syncthreads();
  }

  const float ns = nscale[0];

  // ---------------- stage 2: h x qkw per c' (c=0 then c=2) ----------------
#pragma unroll
  for (int cp = 0; cp < 2; ++cp) {
    // write gelu(h) for this c' into per-wave LDS, row-major bf16 (pitch 136)
    __bf16* hbuf = (__bf16*)lds.h[wave];
#pragma unroll
    for (int nt = 0; nt < 8; ++nt) {
      v8f t = acc[cp * 8 + nt];
#pragma unroll
      for (int v = 0; v < 8; ++v) {
        float x = t[v];
        float g = 0.5f * x * (1.0f + erff(x * 0.70710678118654752f));
        int row = v + ((lane >> 4) << 3);
        int col = nt * 16 + (lane & 15);
        hbuf[row * 136 + col] = (__bf16)g;
      }
    }
    asm volatile("s_wait_dscnt 0x0" ::: "memory");

    v8f acc2[8];
#pragma unroll
    for (int i = 0; i < 8; ++i) acc2[i] = vzero;
#pragma unroll
    for (int kt2 = 0; kt2 < 4; ++kt2) {
      Frag af2;
      const unsigned* hr = &lds.h[wave][(lane & 15) * 68 + kt2 * 16 + ((lane >> 4) << 3)];
      af2.q[0] = *(const uint4*)(hr);
      af2.q[1] = *(const uint4*)(hr + 4);
#pragma unroll
      for (int nt = 0; nt < 8; ++nt) {
        Frag bf2;
        const uint4* qb =
            (const uint4*)(qpack + (size_t)((((cp * 4 + kt2) * 8 + nt) * 32 + lane)) * 8);
        bf2.q[0] = qb[0];
        bf2.q[1] = qb[1];
        acc2[nt] = __builtin_amdgcn_wmma_f32_16x16x32_bf16(
            false, af2.v, false, bf2.v, (short)0, acc2[nt], false, false);
      }
    }
    asm volatile("s_wait_dscnt 0x0" ::: "memory");

    // RMS-norm over m (32 cols = 2 adjacent tiles), scale i>=2, store
#pragma unroll
    for (int i = 0; i < 4; ++i) {
#pragma unroll
      for (int v = 0; v < 8; ++v) {
        float a0 = acc2[2 * i][v];
        float a1 = acc2[2 * i + 1][v];
        float ss = a0 * a0 + a1 * a1;
        ss += __shfl_xor(ss, 1, 32);
        ss += __shfl_xor(ss, 2, 32);
        ss += __shfl_xor(ss, 4, 32);
        ss += __shfl_xor(ss, 8, 32);   // reduce within 16-lane half (cols)
        float sc = rsqrtf(ss * (1.0f / 32.0f) + 1.1920929e-07f);
        if (i >= 2) sc *= ns;
        int row = rowBase + wave * 16 + v + ((lane >> 4) << 3);
        int cb  = cp * 160 + i * 32 + (lane & 15);
        out[(size_t)row * 320 + cb]      = a0 * sc;
        out[(size_t)row * 320 + cb + 16] = a1 * sc;
      }
    }
  }

  // ---------------- dd tail: tanh on packed cols 256..319 ----------------
#pragma unroll
  for (int nt = 16; nt < NT1; ++nt) {
    v8f t = acc[nt];
#pragma unroll
    for (int v = 0; v < 8; ++v) {
      int jcol   = (nt - 16) * 16 + (lane & 15);           // 0..63
      int outcol = (jcol < 32) ? (128 + jcol) : (256 + jcol);
      int row = rowBase + wave * 16 + v + ((lane >> 4) << 3);
      out[(size_t)row * 320 + outcol] = tanhf(t[v]);
    }
  }
}

// ---------------------------------------------------------------------------
extern "C" void kernel_launch(void* const* d_in, const int* in_sizes, int n_in,
                              void* d_out, int out_size, void* d_ws, size_t ws_size,
                              hipStream_t stream) {
  const float* q   = (const float*)d_in[0];   // (4,4096,4096)
  const float* dw1 = (const float*)d_in[1];   // (4096,1,4,128)
  const float* qkw = (const float*)d_in[2];   // (1,4,128,4,32)
  const float* dd  = (const float*)d_in[3];   // (4096,1,128)
  const float* ns  = (const float*)d_in[4];   // (1,)
  float* out = (float*)d_out;

  unsigned* bpack = (unsigned*)d_ws;
  unsigned* qpack = bpack + BPACK_DW;

  pack_b_kernel<<<BPACK_DW / 256, 256, 0, stream>>>(dw1, dd, bpack);
  pack_q_kernel<<<QPACK_DW / 256, 256, 0, stream>>>(qkw, qpack);
  dwp_main_kernel<<<ROWS / 64, 128, 0, stream>>>(q, bpack, qpack, ns, out);
}
